// SinglePermutationLayer_88244398064100
// MI455X (gfx1250) — matrix operand order repaired and checked by reading, not compiled, not run on hardware
//
#include <hip/hip_runtime.h>
#include <hip/hip_bf16.h>
#include <math.h>
#include <stdint.h>

// ---------------------------------------------------------------------------
// MI455X (gfx1250) implementation.  B=64, F=256, S=1024, H=64.
// - bf16 WMMA (fp32 accum) for all GEMMs
// - 8-batch chunks keep the 32MB log-alpha block L2-resident for Sinkhorn
// - final col-normalize fuses exp() + bf16 conversion -> perm buffer
// - permute_kernel: double-buffered global_load_async_to_lds_b128 staging
// ---------------------------------------------------------------------------

#define B_   64
#define F_   256
#define S_   1024
#define H_   64
#define CHUNK 8

typedef __attribute__((ext_vector_type(16))) __bf16 v16bf;
typedef __attribute__((ext_vector_type(8)))  __bf16 v8bf;
typedef __attribute__((ext_vector_type(8)))  float  v8f;

__device__ __forceinline__ v8f wmma_bf16(v16bf a, v16bf b, v8f c) {
  // (neg_a, A, neg_b, B, c_mod, C, reuse_a, reuse_b)
  return __builtin_amdgcn_wmma_f32_16x16x32_bf16(false, a, false, b,
                                                 (short)0, c, false, false);
}

// Fragment layouts (wave32):
//  A (16x32 bf16): lane L -> row rc=L&15, half=L>>4;
//     elems 0..15 -> k = half*16+e (same order as B)
//  B (32x16 bf16): lane L -> col rc; elems 0..15 -> k = half*16+e
//  C/D (16x16 f32): VGPR r, lane L -> row m = r+8*half, col n = rc

__device__ __forceinline__ v16bf frag_from_runs(v8bf lo, v8bf hi) {
  v16bf a;
#pragma unroll
  for (int e = 0; e < 8; e++) { a[e] = lo[e]; a[e + 8] = hi[e]; }
  return a;
}

// ===========================================================================
// Kernel 1: feats = relu(x^T W1 + b1) W2 + b2
// LDS-staged: W1^T/W2^T bf16 once per block; x k-tile transposed per K-step.
// ===========================================================================
__global__ __launch_bounds__(128) void feats_kernel(
    const float* __restrict__ x, const float* __restrict__ W1,
    const float* __restrict__ b1, const float* __restrict__ W2,
    const float* __restrict__ b2, float* __restrict__ feats,
    __bf16* __restrict__ featsb)
{
  __shared__ __align__(16) __bf16 w1t[64 * 264];  // [j][k=0..255], rows 16B-aligned
  __shared__ __align__(16) __bf16 w2t[64 * 72];   // [j][k=0..63]
  __shared__ __align__(16) __bf16 xt [64 * 40];   // [s_local][k=0..31]
  __shared__ float hlds[64 * H_];                 // relu(h) tile

  const int tid  = threadIdx.x;
  const int lane = tid & 31;
  const int wave = tid >> 5;
  const int half = lane >> 4;
  const int rc   = lane & 15;
  const int b    = blockIdx.x / (S_ / 64);
  const int sblk = (blockIdx.x % (S_ / 64)) * 64;
  const int s0   = sblk + wave * 16;

  // ---- stage W1^T, W2^T (bf16, transposed to [col][k]) ----
  for (int i = 0; i < (F_ * H_) / 128; i++) {
    int idx = tid + i * 128;
    int f = idx >> 6, j = idx & 63;                 // consecutive j: coalesced
    w1t[j * 264 + f] = (__bf16)W1[f * H_ + j];
  }
  for (int i = 0; i < (H_ * H_) / 128; i++) {
    int idx = tid + i * 128;
    int k = idx >> 6, j = idx & 63;
    w2t[j * 72 + k] = (__bf16)W2[k * H_ + j];
  }

  // ---- GEMM1: h[s, j] = sum_f x[b, f, s] * W1[f, j] ----
  v8f c[4];
#pragma unroll
  for (int jt = 0; jt < 4; jt++) {
    float bv = b1[jt * 16 + rc];
#pragma unroll
    for (int r = 0; r < 8; r++) c[jt][r] = bv;
  }
  const size_t xbase = (size_t)b * F_ * S_;
  for (int kb = 0; kb < F_ / 32; kb++) {
    __syncthreads();                                // xt free / W-tiles ready
#pragma unroll
    for (int i = 0; i < 16; i++) {                  // coalesced x rows -> xt^T
      int idx = tid + i * 128;
      int r = idx >> 6, scol = idx & 63;
      xt[scol * 40 + r] =
          (__bf16)x[xbase + (size_t)(kb * 32 + r) * S_ + sblk + scol];
    }
    __syncthreads();
    const __bf16* pa = &xt[(wave * 16 + rc) * 40 + half * 16];
    v16bf a = frag_from_runs(*(const v8bf*)pa, *(const v8bf*)(pa + 8));
#pragma unroll
    for (int jt = 0; jt < 4; jt++) {
      const __bf16* pb = &w1t[(jt * 16 + rc) * 264 + kb * 32 + half * 16];
      v16bf bm = frag_from_runs(*(const v8bf*)pb, *(const v8bf*)(pb + 8));
      c[jt] = wmma_bf16(a, bm, c[jt]);
    }
  }
  // relu -> LDS (C->A cross-lane relayout)
#pragma unroll
  for (int jt = 0; jt < 4; jt++)
#pragma unroll
    for (int r = 0; r < 8; r++) {
      float v = c[jt][r];
      hlds[(wave * 16 + r + half * 8) * H_ + jt * 16 + rc] = v > 0.f ? v : 0.f;
    }
  __syncthreads();

  // ---- GEMM2: feats = h W2 + b2 ----
  v8f d[4];
#pragma unroll
  for (int jt = 0; jt < 4; jt++) {
    float bv = b2[jt * 16 + rc];
#pragma unroll
    for (int r = 0; r < 8; r++) d[jt][r] = bv;
  }
#pragma unroll
  for (int kb = 0; kb < H_ / 32; kb++) {
    v16bf a;
#pragma unroll
    for (int e = 0; e < 8; e++) {
      int k0 = kb * 32 + half * 16 + e;
      a[e]     = (__bf16)hlds[(wave * 16 + rc) * H_ + k0];
      a[e + 8] = (__bf16)hlds[(wave * 16 + rc) * H_ + k0 + 8];
    }
#pragma unroll
    for (int jt = 0; jt < 4; jt++) {
      const __bf16* pb = &w2t[(jt * 16 + rc) * 72 + kb * 32 + half * 16];
      v16bf bm = frag_from_runs(*(const v8bf*)pb, *(const v8bf*)(pb + 8));
      d[jt] = wmma_bf16(a, bm, d[jt]);
    }
  }
  const size_t fb = ((size_t)b * S_ + s0) * H_;
#pragma unroll
  for (int jt = 0; jt < 4; jt++)
#pragma unroll
    for (int r = 0; r < 8; r++) {
      float v = d[jt][r];
      size_t idx = fb + (size_t)(r + half * 8) * H_ + jt * 16 + rc;
      feats[idx]  = v;
      featsb[idx] = (__bf16)v;
    }
}

// ===========================================================================
// Kernel 2: log_alpha[bl, s, t] = feats[b,s,:] . feats[b,t,:]  (chunk of 8)
// ===========================================================================
__global__ __launch_bounds__(128) void sim_kernel(
    const __bf16* __restrict__ featsb, float* __restrict__ la, int b0)
{
  const int lane = threadIdx.x & 31;
  const int wave = threadIdx.x >> 5;
  const int half = lane >> 4;
  const int rc   = lane & 15;
  const int bl   = blockIdx.z;
  const int s0   = blockIdx.x * 64 + wave * 16;
  const int t0   = blockIdx.y * 64;
  const __bf16* fb = featsb + (size_t)(b0 + bl) * S_ * H_;

  v8f c[4] = {};
#pragma unroll
  for (int kb = 0; kb < H_ / 32; kb++) {
    const __bf16* ra = fb + (size_t)(s0 + rc) * H_ + kb * 32 + half * 16;
    v16bf a = frag_from_runs(*(const v8bf*)ra, *(const v8bf*)(ra + 8));
#pragma unroll
    for (int nt = 0; nt < 4; nt++) {
      const __bf16* rb = fb + (size_t)(t0 + nt * 16 + rc) * H_ + kb * 32 + half * 16;
      v16bf bm = frag_from_runs(*(const v8bf*)rb, *(const v8bf*)(rb + 8));
      c[nt] = wmma_bf16(a, bm, c[nt]);
    }
  }
  float* out = la + ((size_t)bl * S_ + s0) * S_ + t0;
#pragma unroll
  for (int nt = 0; nt < 4; nt++)
#pragma unroll
    for (int r = 0; r < 8; r++)
      out[(size_t)(r + half * 8) * S_ + nt * 16 + rc] = c[nt][r];
}

// ===========================================================================
// Kernels 3/4: in-place logsumexp-normalize rows (axis 2) / cols (axis 1).
// Values held in registers between passes; chunk is L2-resident.
// ===========================================================================
__global__ __launch_bounds__(256) void lse_row_kernel(float* __restrict__ la)
{
  __shared__ float red[256];
  float* p = la + (size_t)blockIdx.x * S_;
  float v[4], mx = -INFINITY;
#pragma unroll
  for (int i = 0; i < 4; i++) { v[i] = p[threadIdx.x + i * 256]; mx = fmaxf(mx, v[i]); }
  red[threadIdx.x] = mx; __syncthreads();
  for (int st = 128; st > 0; st >>= 1) {
    if (threadIdx.x < st) red[threadIdx.x] = fmaxf(red[threadIdx.x], red[threadIdx.x + st]);
    __syncthreads();
  }
  mx = red[0]; __syncthreads();
  float sum = 0.f;
#pragma unroll
  for (int i = 0; i < 4; i++) sum += __expf(v[i] - mx);
  red[threadIdx.x] = sum; __syncthreads();
  for (int st = 128; st > 0; st >>= 1) {
    if (threadIdx.x < st) red[threadIdx.x] += red[threadIdx.x + st];
    __syncthreads();
  }
  float lse = mx + __logf(red[0]);
#pragma unroll
  for (int i = 0; i < 4; i++) p[threadIdx.x + i * 256] = v[i] - lse;
}

__global__ __launch_bounds__(256) void lse_col_kernel(float* __restrict__ la)
{
  __shared__ float red[256];
  const int bl = blockIdx.x / S_;
  const int t  = blockIdx.x % S_;
  float* p = la + (size_t)bl * S_ * S_ + t;
  float v[4], mx = -INFINITY;
#pragma unroll
  for (int i = 0; i < 4; i++) {
    v[i] = p[(size_t)(threadIdx.x + i * 256) * S_];
    mx = fmaxf(mx, v[i]);
  }
  red[threadIdx.x] = mx; __syncthreads();
  for (int st = 128; st > 0; st >>= 1) {
    if (threadIdx.x < st) red[threadIdx.x] = fmaxf(red[threadIdx.x], red[threadIdx.x + st]);
    __syncthreads();
  }
  mx = red[0]; __syncthreads();
  float sum = 0.f;
#pragma unroll
  for (int i = 0; i < 4; i++) sum += __expf(v[i] - mx);
  red[threadIdx.x] = sum; __syncthreads();
  for (int st = 128; st > 0; st >>= 1) {
    if (threadIdx.x < st) red[threadIdx.x] += red[threadIdx.x + st];
    __syncthreads();
  }
  float lse = mx + __logf(red[0]);
#pragma unroll
  for (int i = 0; i < 4; i++) p[(size_t)(threadIdx.x + i * 256) * S_] = v[i] - lse;
}

// Final col-normalize: perm = exp(log_alpha - lse) as bf16 (T = 1.0).
// Fuses the Sinkhorn exit exp() and the GEMM input conversion in one pass.
__global__ __launch_bounds__(256) void lse_col_final_kernel(
    const float* __restrict__ la, __bf16* __restrict__ permb)
{
  __shared__ float red[256];
  const int bl = blockIdx.x / S_;
  const int t  = blockIdx.x % S_;
  const float* p = la + (size_t)bl * S_ * S_ + t;
  __bf16* q = permb + (size_t)bl * S_ * S_ + t;
  float v[4], mx = -INFINITY;
#pragma unroll
  for (int i = 0; i < 4; i++) {
    v[i] = p[(size_t)(threadIdx.x + i * 256) * S_];
    mx = fmaxf(mx, v[i]);
  }
  red[threadIdx.x] = mx; __syncthreads();
  for (int st = 128; st > 0; st >>= 1) {
    if (threadIdx.x < st) red[threadIdx.x] = fmaxf(red[threadIdx.x], red[threadIdx.x + st]);
    __syncthreads();
  }
  mx = red[0]; __syncthreads();
  float sum = 0.f;
#pragma unroll
  for (int i = 0; i < 4; i++) sum += __expf(v[i] - mx);
  red[threadIdx.x] = sum; __syncthreads();
  for (int st = 128; st > 0; st >>= 1) {
    if (threadIdx.x < st) red[threadIdx.x] += red[threadIdx.x + st];
    __syncthreads();
  }
  float lse = mx + __logf(red[0]);
#pragma unroll
  for (int i = 0; i < 4; i++)
    q[(size_t)(threadIdx.x + i * 256) * S_] = (__bf16)__expf(v[i] - lse);
}

// ===========================================================================
// Kernel 5: out[b, f, s] = sum_t perm[bl, s, t] * x[b, f, t]
// Double-buffered global_load_async_to_lds_b128 staging of x tiles: the DMA
// for step kb+1 runs under the WMMA work of step kb (s_wait_asynccnt 4).
// ===========================================================================
__device__ __forceinline__ void issue_xtile_dma(
    const float* __restrict__ xb, int kb, uint32_t ldsoff, int tid)
{
#pragma unroll
  for (int i = 0; i < 4; i++) {
    int cidx = tid + i * 128;
    int r = cidx >> 3, j = cidx & 7;
    const float* g = xb + (size_t)r * S_ + kb * 32 + j * 4;
    uint32_t loff = ldsoff + (uint32_t)(r * 144 + j * 16);
    asm volatile("global_load_async_to_lds_b128 %0, %1, off"
                 :: "v"(loff), "v"((uint64_t)(uintptr_t)g)
                 : "memory");
  }
}

__global__ __launch_bounds__(128) void permute_kernel(
    const __bf16* __restrict__ permb, const float* __restrict__ x,
    float* __restrict__ out, int b0)
{
  __shared__ __align__(16) float xtile[2][64 * 36];  // [f_local][k], 144B rows
  const int tid  = threadIdx.x;
  const int lane = tid & 31;
  const int wave = tid >> 5;
  const int half = lane >> 4;
  const int rc   = lane & 15;
  const int bl   = blockIdx.z;
  const int b    = b0 + bl;
  const int s0   = blockIdx.x * 64 + wave * 16;
  const int f0   = blockIdx.y * 64;
  const __bf16* prow = permb + ((size_t)bl * S_ + s0 + rc) * S_;
  const float*  xb   = x + ((size_t)b * F_ + f0) * S_;
  const uint32_t lds0 = (uint32_t)(uintptr_t)(&xtile[0][0]);
  const uint32_t lds1 = (uint32_t)(uintptr_t)(&xtile[1][0]);

  v8f c[4] = {};
  issue_xtile_dma(xb, 0, lds0, tid);                 // prologue: tile 0 -> buf 0
  for (int kb = 0; kb < S_ / 32; kb++) {
    const int cur = kb & 1;
    __syncthreads();            // all waves done reading buf[cur^1] (iter kb-1)
    if (kb + 1 < S_ / 32)
      issue_xtile_dma(xb, kb + 1, cur ? lds0 : lds1, tid);
    // A fragment: bf16 perm rows, two 16B runs (overlaps in-flight DMA)
    const __bf16* pa = prow + kb * 32 + half * 16;
    v16bf a = frag_from_runs(*(const v8bf*)pa, *(const v8bf*)(pa + 8));
    if (kb + 1 < S_ / 32)
      asm volatile("s_wait_asynccnt 0x4" ::: "memory");  // batch kb complete
    else
      asm volatile("s_wait_asynccnt 0x0" ::: "memory");
    __syncthreads();
#pragma unroll
    for (int nt = 0; nt < 4; nt++) {
      const float* pb = &xtile[cur][(nt * 16 + rc) * 36 + half * 16];
      float tb[16];
      *(float4*)(tb + 0)  = *(const float4*)(pb + 0);
      *(float4*)(tb + 4)  = *(const float4*)(pb + 4);
      *(float4*)(tb + 8)  = *(const float4*)(pb + 8);
      *(float4*)(tb + 12) = *(const float4*)(pb + 12);
      v16bf bm;
#pragma unroll
      for (int e = 0; e < 16; e++) bm[e] = (__bf16)tb[e];
      c[nt] = wmma_bf16(a, bm, c[nt]);
    }
  }
#pragma unroll
  for (int nt = 0; nt < 4; nt++) {
    float* po = out + ((size_t)b * F_ + f0 + nt * 16 + rc) * S_ + s0 + half * 8;
    *(float4*)(po + 0) = make_float4(c[nt][0], c[nt][1], c[nt][2], c[nt][3]);
    *(float4*)(po + 4) = make_float4(c[nt][4], c[nt][5], c[nt][6], c[nt][7]);
  }
}

// ===========================================================================
extern "C" void kernel_launch(void* const* d_in, const int* in_sizes, int n_in,
                              void* d_out, int out_size, void* d_ws, size_t ws_size,
                              hipStream_t stream)
{
  const float* x  = (const float*)d_in[0];
  const float* W1 = (const float*)d_in[1];
  const float* b1 = (const float*)d_in[2];
  const float* W2 = (const float*)d_in[3];
  const float* b2 = (const float*)d_in[4];
  float* out = (float*)d_out;

  // Workspace: feats fp32 (16MB) | feats bf16 (8MB) | log_alpha chunk (32MB)
  //            | perm bf16 chunk (16MB)
  char* ws = (char*)d_ws;
  float*  feats  = (float*)ws;   ws += (size_t)B_ * S_ * H_ * 4;
  __bf16* featsb = (__bf16*)ws;  ws += (size_t)B_ * S_ * H_ * 2;
  float*  la     = (float*)ws;   ws += (size_t)CHUNK * S_ * S_ * 4;
  __bf16* permb  = (__bf16*)ws;

  feats_kernel<<<dim3(B_ * (S_ / 64)), 128, 0, stream>>>(x, W1, b1, W2, b2,
                                                         feats, featsb);

  for (int b0 = 0; b0 < B_; b0 += CHUNK) {
    sim_kernel<<<dim3(S_ / 64, S_ / 64, CHUNK), 128, 0, stream>>>(featsb, la, b0);
    for (int it = 0; it < 5; ++it) {
      lse_row_kernel<<<dim3(CHUNK * S_), 256, 0, stream>>>(la);
      if (it < 4)
        lse_col_kernel<<<dim3(CHUNK * S_), 256, 0, stream>>>(la);
      else
        lse_col_final_kernel<<<dim3(CHUNK * S_), 256, 0, stream>>>(la, permb);
    }
    permute_kernel<<<dim3(S_ / 64, F_ / 64, CHUNK), 128, 0, stream>>>(permb, x, out, b0);
  }
}
